// GeneToPathwayAttention_90993177133726
// MI455X (gfx1250) — matrix operand order repaired and checked by reading, not compiled, not run on hardware
//
#include <hip/hip_runtime.h>
#include <hip/hip_bf16.h>

#define DEVINL __device__ __forceinline__

typedef __attribute__((ext_vector_type(16))) __bf16 v16bf;
typedef __attribute__((ext_vector_type(8)))  __bf16 v8bf;
typedef __attribute__((ext_vector_type(8)))  float  v8f;

static constexpr int   Bn      = 8;
static constexpr int   G       = 10000;
static constexpr int   P       = 1500;
static constexpr int   D       = 128;
static constexpr int   Gpad    = 10016;             // mult of 16, safety pad
static constexpr int   Ppad    = 1504;              // P padded to mult of 16
static constexpr int   Wb      = 313;               // ceil(G/32) mask words per row
static constexpr int   GT      = G / 16;            // 625 g-tiles (exact)
static constexpr int   PT      = (P + 15) / 16;     // 94 p-tiles
static constexpr float kNegInf = -1e9f;

DEVINL v8f wmma_bf16(v16bf a, v16bf b, v8f c) {
  // D = A(16x32 bf16) x B(32x16 bf16) + C(16x16 f32)
  return __builtin_amdgcn_wmma_f32_16x16x32_bf16(false, a, false, b, (short)0, c,
                                                 false, false);
}

DEVINL v16bf load_v16bf(const __bf16* lo, const __bf16* hi) {
  v8bf l = *(const v8bf*)lo;
  v8bf h = *(const v8bf*)hi;
  v16bf r;
#pragma unroll
  for (int i = 0; i < 8; ++i) { r[i] = l[i]; r[i + 8] = h[i]; }
  return r;
}

// ---------------------------------------------------------------- prep kernels
__global__ void prep_mask_kernel(const int* __restrict__ A, unsigned* __restrict__ Abits) {
  int idx = blockIdx.x * blockDim.x + threadIdx.x;
  if (idx >= Ppad * Wb) return;
  int p = idx / Wb, w = idx % Wb;
  unsigned bits = 0;
  if (p < P) {
    int gbase = w * 32;
#pragma unroll 4
    for (int i = 0; i < 32; ++i) {
      int g = gbase + i;
      if (g < G && A[(size_t)p * G + g] != 0) bits |= (1u << i);
    }
  }
  Abits[idx] = bits;  // pad rows -> all masked
}

__global__ void prep_q_kernel(const float* __restrict__ Q, __bf16* __restrict__ Qbf) {
  int idx = blockIdx.x * blockDim.x + threadIdx.x;
  if (idx >= Ppad * D) return;
  int p = idx / D;
  Qbf[idx] = (p < P) ? (__bf16)Q[idx] : (__bf16)0.0f;
}

// ------------------------------------------------------------- K/V projection
// K[b,g,:] = gene[b,g,:] @ Wk^T + bk  (row-major bf16)
// V stored transposed: Vt[b,d,g]     (bf16) for contiguous B-fragments later.
__global__ __launch_bounds__(256) void kv_proj_kernel(
    const float* __restrict__ X,            // gene_emb as [B*G, 128]
    const float* __restrict__ Wk, const float* __restrict__ bk,
    const float* __restrict__ Wv, const float* __restrict__ bv,
    __bf16* __restrict__ Kb,                // [B][Gpad][128]
    __bf16* __restrict__ Vt)                // [B][128][Gpad]
{
  const int tile = blockIdx.x;              // 16 gene rows per tile; G%16==0
  const int wave = threadIdx.x >> 5;
  const int lane = threadIdx.x & 31;
  const int n = lane & 15, half = lane >> 4;

  const size_t row0 = (size_t)tile * 16;
  // A fragments: gene rows, fp32 -> bf16.  m = lane&15; K = j + 16*half
  const float* xrow = X + (row0 + n) * D + 16 * half;
  v16bf afrag[4];
#pragma unroll
  for (int c = 0; c < 4; ++c) {
    const float* pa = xrow + 32 * c;
    v16bf a;
#pragma unroll
    for (int i = 0; i < 8; ++i) {
      a[i]     = (__bf16)pa[i];
      a[i + 8] = (__bf16)pa[8 + i];
    }
    afrag[c] = a;
  }

  const int b  = (int)(row0 / G);
  const int g0 = (int)(row0 % G);
  const int nt = wave;                      // each wave owns one 16-col n-tile

  // B fragment: B[k][n] = W[nt*16+n][k]; lane n, K = 16*half + j
  v8f acc;

  // ---- K projection (compile-time branch-free store path) ----
  {
    const float* wrow = Wk + (size_t)(nt * 16 + n) * D + 16 * half;
    v8f c = {};
#pragma unroll
    for (int cc = 0; cc < 4; ++cc) {
      const float* pb = wrow + 32 * cc;
      v16bf bfrag;
#pragma unroll
      for (int i = 0; i < 16; ++i) bfrag[i] = (__bf16)pb[i];
      c = wmma_bf16(afrag[cc], bfrag, c);
    }
    const float bias = bk[nt * 16 + n];
#pragma unroll
    for (int r = 0; r < 8; ++r) {           // D row M = r + 8*half, N = n
      const int g = g0 + r + 8 * half;
      Kb[((size_t)b * Gpad + g) * D + nt * 16 + n] = (__bf16)(c[r] + bias);
    }
    acc = c;
    (void)acc;
  }

  // ---- V projection ----
  {
    const float* wrow = Wv + (size_t)(nt * 16 + n) * D + 16 * half;
    v8f c = {};
#pragma unroll
    for (int cc = 0; cc < 4; ++cc) {
      const float* pb = wrow + 32 * cc;
      v16bf bfrag;
#pragma unroll
      for (int i = 0; i < 16; ++i) bfrag[i] = (__bf16)pb[i];
      c = wmma_bf16(afrag[cc], bfrag, c);
    }
    const float bias = bv[nt * 16 + n];
#pragma unroll
    for (int r = 0; r < 8; ++r) {
      const int g = g0 + r + 8 * half;
      Vt[((size_t)b * D + nt * 16 + n) * Gpad + g] = (__bf16)(c[r] + bias);
    }
  }
}

// ------------------------------------------------------------ attention kernel
DEVINL void logits_tile(const __bf16* __restrict__ Kb,
                        const unsigned* s_mask,    // LDS: [16][Wb] for this p-tile
                        const v16bf qfrag[4],
                        int g0, int n, int half, float scale,
                        float s[8])
{
  // B fragment of K^T: B[k][g'] = K[g0+g'][k]; lane n = g', K = 16*half + j
  const __bf16* krow = Kb + (size_t)(g0 + n) * D + 16 * half;
  v8f c = {};
#pragma unroll
  for (int cc = 0; cc < 4; ++cc) {
    v16bf bf = load_v16bf(krow + 32 * cc, krow + 32 * cc + 8);
    c = wmma_bf16(qfrag[cc], bf, c);
  }
  const int g = g0 + n;
  const unsigned w = ((unsigned)g) >> 5, bit = (unsigned)g & 31u;
#pragma unroll
  for (int r = 0; r < 8; ++r) {
    const int row = r + 8 * half;
    const unsigned aw = s_mask[row * Wb + w];
    s[r] = ((aw >> bit) & 1u) ? c[r] * scale : kNegInf;
  }
}

__global__ __launch_bounds__(256) void attn_kernel(
    const __bf16* __restrict__ Kball,       // [B][Gpad][128]
    const __bf16* __restrict__ Vtall,       // [B][128][Gpad]
    const __bf16* __restrict__ Qbf,         // [Ppad][128]
    const unsigned* __restrict__ Abits,     // [Ppad][Wb]
    float* __restrict__ out_pw,             // [B][P][128]
    float* __restrict__ out_attn)           // [B][P][G]
{
  const int p0 = blockIdx.x * 16;
  const int b  = blockIdx.y;
  const int wave = threadIdx.x >> 5, lane = threadIdx.x & 31;
  const int n = lane & 15, half = lane >> 4;

  __shared__ unsigned s_mask[16 * Wb];      // packed mask rows for this p-tile
  __shared__ float    s_O[16 * 128];
  __shared__ float    s_m[8][16];
  __shared__ float    s_l[8][16];
  __shared__ __bf16   s_attn[8][16 * 32];

  for (int i = threadIdx.x; i < 16 * 128; i += 256) s_O[i] = 0.0f;
  for (int i = threadIdx.x; i < 16 * Wb; i += 256) {
    const int row = i / Wb, w = i - row * Wb;
    s_mask[i] = Abits[(size_t)(p0 + row) * Wb + w];
  }

  const __bf16* Kb = Kball + (size_t)b * Gpad * D;
  const __bf16* Vt = Vtall + (size_t)b * D * Gpad;
  const float scale = 0.08838834764831845f;  // 1/sqrt(128)

  // Q fragments live in registers for the whole kernel (4 x v16bf)
  const __bf16* qrow = Qbf + (size_t)(p0 + n) * D + 16 * half;
  v16bf qfrag[4];
#pragma unroll
  for (int c = 0; c < 4; ++c)
    qfrag[c] = load_v16bf(qrow + 32 * c, qrow + 32 * c + 8);

  __syncthreads();  // s_mask ready

  // -------- pass 1: per-lane online softmax stats (no shuffles in loop) ------
  float mrun[8], lrun[8];
#pragma unroll
  for (int r = 0; r < 8; ++r) { mrun[r] = -INFINITY; lrun[r] = 0.0f; }

  for (int t = wave; t < GT; t += 8) {
    if (t + 8 < GT)  // prefetch next K tile row (global_prefetch_b8)
      __builtin_prefetch(Kb + (size_t)((t + 8) * 16 + n) * D, 0, 0);
    float s[8];
    logits_tile(Kb, s_mask, qfrag, t * 16, n, half, scale, s);
#pragma unroll
    for (int r = 0; r < 8; ++r) {
      const float sv = s[r];
      if (sv > mrun[r]) {                    // exactly one exp either way
        lrun[r] = lrun[r] * __expf(mrun[r] - sv) + 1.0f;
        mrun[r] = sv;
      } else {
        lrun[r] += __expf(sv - mrun[r]);
      }
    }
  }
  // butterfly merge of (m,l) across the 16 lanes of each half (once)
#pragma unroll
  for (int r = 0; r < 8; ++r) {
    float m = mrun[r], l = lrun[r];
#pragma unroll
    for (int xm = 1; xm < 16; xm <<= 1) {
      const float m2 = __shfl_xor(m, xm, 32);
      const float l2 = __shfl_xor(l, xm, 32);
      const float mn = fmaxf(m, m2);
      l = l * __expf(m - mn) + l2 * __expf(m2 - mn);
      m = mn;
    }
    mrun[r] = m; lrun[r] = l;
  }
  if (n == 0) {
#pragma unroll
    for (int r = 0; r < 8; ++r) {
      s_m[wave][r + 8 * half] = mrun[r];
      s_l[wave][r + 8 * half] = lrun[r];
    }
  }
  __syncthreads();

  float Mrow[8], invL[8];
#pragma unroll
  for (int r = 0; r < 8; ++r) {
    const int row = r + 8 * half;
    float m = -INFINITY;
#pragma unroll
    for (int w2 = 0; w2 < 8; ++w2) m = fmaxf(m, s_m[w2][row]);
    float l = 0.0f;
#pragma unroll
    for (int w2 = 0; w2 < 8; ++w2) l += s_l[w2][row] * __expf(s_m[w2][row] - m);
    Mrow[r] = m;
    invL[r] = 1.0f / l;
  }

  // -------- pass 2: attn write (once) + attn@V accumulate --------
  v8f Oacc[8];
#pragma unroll
  for (int i = 0; i < 8; ++i) { v8f z = {}; Oacc[i] = z; }

  __bf16* satt = &s_attn[wave][0];
  for (int t = wave; t < GT; t += 16) {          // pair (t, t+8) -> K=32
    const int tB = t + 8;
    const bool hasB = tB < GT;
#pragma unroll
    for (int side = 0; side < 2; ++side) {
      const int tt = side ? tB : t;
      if (side && !hasB) {
#pragma unroll
        for (int r = 0; r < 8; ++r)
          satt[(r + 8 * half) * 32 + 16 + n] = (__bf16)0.0f;
        continue;
      }
      float s[8];
      logits_tile(Kb, s_mask, qfrag, tt * 16, n, half, scale, s);
#pragma unroll
      for (int r = 0; r < 8; ++r) {
        const int p = p0 + r + 8 * half;
        const float pv = __expf(s[r] - Mrow[r]) * invL[r];
        satt[(r + 8 * half) * 32 + side * 16 + n] = (__bf16)pv;
        if (p < P) out_attn[(size_t)(b * P + p) * G + tt * 16 + n] = pv;
      }
    }
    // A' fragment (attn tile, M=p, K=g over 32) from per-wave LDS
    v16bf af;
#pragma unroll
    for (int j = 0; j < 16; ++j) {
      const int kk = j + 16 * half;
      af[j] = satt[n * 32 + kk];
    }
    // B fragment of V: B[k][d'] = V[g(k)][d0+d'] = Vt[d0+d'][g(k)]
    const int gbase = (half ? (hasB ? tB : t) : t) * 16;
#pragma unroll
    for (int nt = 0; nt < 8; ++nt) {
      const __bf16* vrow = Vt + (size_t)(nt * 16 + n) * Gpad + gbase;
      v16bf vf = load_v16bf(vrow, vrow + 8);
      Oacc[nt] = wmma_bf16(af, vf, Oacc[nt]);
    }
  }

  // cross-wave reduce O in LDS, then store pw_emb
#pragma unroll
  for (int nt = 0; nt < 8; ++nt)
#pragma unroll
    for (int r = 0; r < 8; ++r)
      atomicAdd(&s_O[(r + 8 * half) * 128 + nt * 16 + n], Oacc[nt][r]);
  __syncthreads();
  for (int i = threadIdx.x; i < 16 * 128; i += 256) {
    const int row = i >> 7, d = i & 127;
    const int p = p0 + row;
    if (p < P) out_pw[(size_t)(b * P + p) * 128 + d] = s_O[i];
  }
}

// -------------------------------------------------------------------- launcher
extern "C" void kernel_launch(void* const* d_in, const int* in_sizes, int n_in,
                              void* d_out, int out_size, void* d_ws, size_t ws_size,
                              hipStream_t stream) {
  const float* gene = (const float*)d_in[0];
  const int*   A    = (const int*)d_in[1];
  const float* Wk   = (const float*)d_in[2];
  const float* bk   = (const float*)d_in[3];
  const float* Wv   = (const float*)d_in[4];
  const float* bv   = (const float*)d_in[5];
  const float* Q    = (const float*)d_in[6];

  // workspace layout (bytes): K bf16 | Vt bf16 | Q bf16 | mask bits  (~43.3 MB)
  char* ws = (char*)d_ws;
  size_t off = 0;
  __bf16* Kb  = (__bf16*)(ws + off); off += (size_t)Bn * Gpad * D * 2;
  __bf16* Vt  = (__bf16*)(ws + off); off += (size_t)Bn * D * Gpad * 2;
  __bf16* Qbf = (__bf16*)(ws + off); off += (size_t)Ppad * D * 2;
  unsigned* Abits = (unsigned*)(ws + off);

  float* out_pw   = (float*)d_out;
  float* out_attn = out_pw + (size_t)Bn * P * D;

  {
    int tot = Ppad * Wb;
    prep_mask_kernel<<<(tot + 255) / 256, 256, 0, stream>>>(A, Abits);
  }
  {
    int tot = Ppad * D;
    prep_q_kernel<<<(tot + 255) / 256, 256, 0, stream>>>(Q, Qbf);
  }
  kv_proj_kernel<<<(Bn * G) / 16, 256, 0, stream>>>(gene, Wk, bk, Wv, bv, Kb, Vt);
  attn_kernel<<<dim3(PT, Bn), 256, 0, stream>>>(Kb, Vt, Qbf, Abits, out_pw, out_attn);
}
